// SparseLeastSquares_66176856096949
// MI455X (gfx1250) — matrix operand order repaired and checked by reading, not compiled, not run on hardware
//
#include <hip/hip_runtime.h>
#include <hip/hip_bf16.h>

// Problem constants (match reference)
#define D_SAE 16384
#define D_IN  4096
#define SEQ   4096
#define KSUP  512

typedef __attribute__((ext_vector_type(8)))  float   v8f;
typedef __attribute__((ext_vector_type(16))) __bf16  v16bf;

// ---------------------------------------------------------------------------
// 1) Zero the 4096 x 16384 f32 output with wide stores (256 MB -> HBM bound)
// ---------------------------------------------------------------------------
__global__ void zero_out_kernel(float4* __restrict__ out) {
    size_t i = (size_t)blockIdx.x * blockDim.x + threadIdx.x;
    out[i] = make_float4(0.f, 0.f, 0.f, 0.f);
}

// ---------------------------------------------------------------------------
// 2) Gather A = dictionary[support]  (512 x 4096 f32), one block per row
// ---------------------------------------------------------------------------
__global__ void gather_rows_kernel(const float* __restrict__ dict,
                                   const int* __restrict__ support,
                                   float* __restrict__ A) {
    const int k   = blockIdx.x;
    const int row = support[k];
    const float4* src = (const float4*)(dict + (size_t)row * D_IN);
    float4*       dst = (float4*)(A + (size_t)k * D_IN);
    for (int i = threadIdx.x; i < D_IN / 4; i += blockDim.x)
        dst[i] = src[i];
}

// ---------------------------------------------------------------------------
// 3) C[m,n] = dot(X[m,:], Z[n,:]) via split-bf16 WMMA (hi*hi + hi*lo + lo*hi).
//    One wave (32 threads) per 16x64 output strip: the converted A operand is
//    reused across 4 B tiles, and the 12 WMMAs per K-step round-robin over 4
//    independent accumulators so the XDL pipe never stalls on a D->C RAW.
//    Operand layouts follow CDNA5 ISA §7.12.2 (wave32).
// ---------------------------------------------------------------------------
__global__ void __launch_bounds__(32)
gemm_nt_split_bf16(const float* __restrict__ X,  // M x Kdim, row-major, ld = ldx
                   const float* __restrict__ Z,  // N x Kdim, row-major, ld = ldz
                   float* __restrict__ C,        // M x N,   row-major, ld = ldc
                   int ldx, int ldz, int ldc, int Kdim) {
    __shared__ float sA[16][32];   // 2 KB
    __shared__ float sB[64][32];   // 8 KB

    const int lane = threadIdx.x;           // 0..31
    const int m0 = blockIdx.x * 16;
    const int n0 = blockIdx.y * 64;

    // cooperative tile load mapping: 2 lanes per row, 16 contiguous f32 each
    const int lrow = lane >> 1;
    const int lcol = (lane & 1) * 16;

    const int half = lane >> 4;             // 0: lanes 0-15, 1: lanes 16-31
    const int mn   = lane & 15;             // row (A operand) / col (B operand)

    v8f acc[4] = {};

    for (int k0 = 0; k0 < Kdim; k0 += 32) {
        // ---- stage f32 tiles in LDS (coalesced float4) ----
        const float4* pa = (const float4*)(X + (size_t)(m0 + lrow) * ldx + k0 + lcol);
        float4* da = (float4*)(&sA[lrow][lcol]);
#pragma unroll
        for (int q = 0; q < 4; ++q) da[q] = pa[q];
#pragma unroll
        for (int t = 0; t < 4; ++t) {
            const int brow = t * 16 + lrow;
            const float4* pb = (const float4*)(Z + (size_t)(n0 + brow) * ldz + k0 + lcol);
            float4* db = (float4*)(&sB[brow][lcol]);
#pragma unroll
            for (int q = 0; q < 4; ++q) db[q] = pb[q];
        }
        // prefetch next K-step rows (emits global_prefetch on gfx1250)
        if (k0 + 32 < Kdim) {
            __builtin_prefetch(X + (size_t)(m0 + lrow) * ldx + k0 + 32 + lcol, 0, 3);
#pragma unroll
            for (int t = 0; t < 4; ++t)
                __builtin_prefetch(Z + (size_t)(n0 + t * 16 + lrow) * ldz + k0 + 32 + lcol, 0, 3);
        }
        __syncthreads();

        // ---- build A operand once (hi/lo split) ----
        v16bf ahi, alo;
#pragma unroll
        for (int j = 0; j < 16; ++j) {
            // A 16x32 bf16 layout: VGPR j>>1, K = (j&7) + 8*half + 16*(j>>3)
            const int ka = (j & 7) + 8 * half + 16 * (j >> 3);
            const float va = sA[mn][ka];
            const __bf16 ha = (__bf16)va;
            ahi[j] = ha;
            alo[j] = (__bf16)(va - (float)ha);
        }
        // ---- build 4 B operands (hi/lo split) ----
        v16bf bhi[4], blo[4];
#pragma unroll
        for (int t = 0; t < 4; ++t) {
#pragma unroll
            for (int j = 0; j < 16; ++j) {
                // B 32x16 bf16 layout: K = (j&7) + 8*half + 16*(j>>3) (matches A)
                const int kb = (j & 7) + 8 * half + 16 * (j >> 3);
                const float vb = sB[t * 16 + mn][kb];
                const __bf16 hb = (__bf16)vb;
                bhi[t][j] = hb;
                blo[t][j] = (__bf16)(vb - (float)hb);
            }
        }

        // ---- 12 WMMAs: 3 split passes x 4 independent accumulators ----
#pragma unroll
        for (int t = 0; t < 4; ++t)
            acc[t] = __builtin_amdgcn_wmma_f32_16x16x32_bf16(
                false, ahi, false, bhi[t], (short)0, acc[t], false, false);
#pragma unroll
        for (int t = 0; t < 4; ++t)
            acc[t] = __builtin_amdgcn_wmma_f32_16x16x32_bf16(
                false, ahi, false, blo[t], (short)0, acc[t], false, false);
#pragma unroll
        for (int t = 0; t < 4; ++t)
            acc[t] = __builtin_amdgcn_wmma_f32_16x16x32_bf16(
                false, alo, false, bhi[t], (short)0, acc[t], false, false);
        __syncthreads();
    }

    // C/D f32 16x16 layout: VGPR r -> M = r + 8*half, N = lane&15
#pragma unroll
    for (int t = 0; t < 4; ++t) {
#pragma unroll
        for (int r = 0; r < 8; ++r) {
            const int m = r + 8 * half;
            C[(size_t)(m0 + m) * ldc + n0 + t * 16 + mn] = acc[t][r];
        }
    }
}

// ---------------------------------------------------------------------------
// 4) In-place lower Cholesky of G (512x512, SPD). Single block, 256 threads.
// ---------------------------------------------------------------------------
__global__ void cholesky512_kernel(float* __restrict__ G) {
    const int tid = threadIdx.x;
    for (int j = 0; j < KSUP; ++j) {
        if (tid == 0) G[j * KSUP + j] = sqrtf(G[j * KSUP + j]);
        __syncthreads();
        const float d = G[j * KSUP + j];
        for (int i = j + 1 + tid; i < KSUP; i += 256)
            G[i * KSUP + j] /= d;
        __syncthreads();
        // trailing rank-1 update on the lower triangle
        for (int i = j + 1; i < KSUP; ++i) {
            const float gij = G[i * KSUP + j];
            for (int ccol = j + 1 + tid; ccol <= i; ccol += 256)
                G[i * KSUP + ccol] -= gij * G[ccol * KSUP + j];
        }
        __syncthreads();
    }
}

// ---------------------------------------------------------------------------
// 5) Solve (L L^T) X = B in place for 4096 RHS columns of B (512 x 4096).
//    Each block owns 16 RHS columns, staged in 32 KB of LDS.
// ---------------------------------------------------------------------------
__global__ void solve_chol_kernel(const float* __restrict__ G,
                                  float* __restrict__ Bm) {
    __shared__ float xs[KSUP][16];   // 32 KB
    const int tid = threadIdx.x;     // 0..255
    const int c0  = blockIdx.x * 16;
    const int c   = tid & 15;
    const int r0  = tid >> 4;        // 0..15

    for (int k = r0; k < KSUP; k += 16)
        xs[k][c] = Bm[(size_t)k * SEQ + c0 + c];
    __syncthreads();

    // forward: L y = b
    for (int j = 0; j < KSUP; ++j) {
        if (tid < 16) xs[j][tid] /= G[j * KSUP + j];
        __syncthreads();
        const float xj = xs[j][c];
        for (int i = j + 1 + r0; i < KSUP; i += 16)
            xs[i][c] -= G[i * KSUP + j] * xj;
        __syncthreads();
    }
    // backward: L^T x = y
    for (int j = KSUP - 1; j >= 0; --j) {
        if (tid < 16) xs[j][tid] /= G[j * KSUP + j];
        __syncthreads();
        const float xj = xs[j][c];
        for (int i = r0; i < j; i += 16)
            xs[i][c] -= G[j * KSUP + i] * xj;
        __syncthreads();
    }

    for (int k = r0; k < KSUP; k += 16)
        Bm[(size_t)k * SEQ + c0 + c] = xs[k][c];
}

// ---------------------------------------------------------------------------
// 6) Scatter coeff (K x SEQ) into out[s, support[k]]
// ---------------------------------------------------------------------------
__global__ void scatter_kernel(const float* __restrict__ coeff,
                               const int* __restrict__ support,
                               float* __restrict__ out) {
    const int k = blockIdx.x;                       // 0..511
    const int s = blockIdx.y * 256 + threadIdx.x;   // 0..4095
    const int col = support[k];
    out[(size_t)s * D_SAE + col] = coeff[(size_t)k * SEQ + s];
}

// ---------------------------------------------------------------------------
// Host launch
// ---------------------------------------------------------------------------
extern "C" void kernel_launch(void* const* d_in, const int* in_sizes, int n_in,
                              void* d_out, int out_size, void* d_ws, size_t ws_size,
                              hipStream_t stream) {
    const float* dict    = (const float*)d_in[0];   // (D_SAE, D_IN)
    const float* targets = (const float*)d_in[1];   // (SEQ, D_IN)
    const int*   support = (const int*)d_in[2];     // (K,)
    float* out = (float*)d_out;                     // (SEQ, D_SAE)

    // workspace: A32 (8 MB) | G (1 MB) | B/coeff (8 MB)
    float* A32 = (float*)d_ws;
    float* G   = A32 + (size_t)KSUP * D_IN;
    float* Bm  = G   + (size_t)KSUP * KSUP;

    // 1) zero output (64M f32 = 16M float4)
    zero_out_kernel<<<(SEQ * (size_t)D_SAE) / 4 / 256, 256, 0, stream>>>((float4*)out);

    // 2) gather A = dict[support]
    gather_rows_kernel<<<KSUP, 256, 0, stream>>>(dict, support, A32);

    // 3) G = A A^T  (512x512, K=4096)
    gemm_nt_split_bf16<<<dim3(KSUP / 16, KSUP / 64), 32, 0, stream>>>(
        A32, A32, G, D_IN, D_IN, KSUP, D_IN);

    // 4) B = A Y^T  (512x4096, K=4096)
    gemm_nt_split_bf16<<<dim3(KSUP / 16, SEQ / 64), 32, 0, stream>>>(
        A32, targets, Bm, D_IN, D_IN, SEQ, D_IN);

    // 5) Cholesky of G
    cholesky512_kernel<<<1, 256, 0, stream>>>(G);

    // 6) triangular solves, coeff overwrites B
    solve_chol_kernel<<<SEQ / 16, 256, 0, stream>>>(G, Bm);

    // 7) scatter into output
    scatter_kernel<<<dim3(KSUP, SEQ / 256), 256, 0, stream>>>(Bm, support, out);
}
